// Attention_40475771798025
// MI455X (gfx1250) — matrix operand order repaired and checked by reading, not compiled, not run on hardware
//
#include <hip/hip_runtime.h>
#include <hip/hip_bf16.h>

// ---------------------------------------------------------------------------
// Problem constants (B,S,D)=(2,2048,4096), H=32, KV=8, HD=128, REP=4
// ---------------------------------------------------------------------------
#define B_    2
#define S_    2048
#define D_    4096
#define H_    32
#define KV_   8
#define HD_   128
#define REP_  4
#define BS_   (B_ * S_)          // 4096 rows
#define KD_   (KV_ * HD_)        // 1024
#define SCALE_ 0.08838834764831845f  // 1/sqrt(128)

typedef __bf16 bf16_t;
typedef __attribute__((ext_vector_type(16))) __bf16 v16bf;
typedef __attribute__((ext_vector_type(8)))  float  v8f;
typedef unsigned int v4u __attribute__((ext_vector_type(4)));
typedef int          v8i __attribute__((ext_vector_type(8)));
typedef int          v4i __attribute__((ext_vector_type(4)));

#if defined(__gfx1250__) && __has_builtin(__builtin_amdgcn_tensor_load_to_lds)
#define USE_TDM 1
#else
#define USE_TDM 0
#endif

union FragB {              // one WMMA bf16 operand: 16 bf16 = 32B = 2x uint4
    v16bf v;
    uint4 q[2];
    bf16_t e[16];
};

__device__ __forceinline__ v8f wmma_bf16(v16bf a, v16bf b, v8f c) {
    return __builtin_amdgcn_wmma_f32_16x16x32_bf16(
        /*neg_a=*/false, a, /*neg_b=*/false, b,
        /*c_mod=*/(short)0, c, /*reuse_a=*/false, /*reuse_b=*/false);
}

__device__ __forceinline__ v8f v8f_zero() {
    v8f z = {0.f, 0.f, 0.f, 0.f, 0.f, 0.f, 0.f, 0.f};
    return z;
}

#if USE_TDM
// ---------------------------------------------------------------------------
// Tensor Data Mover: 2D tile load (global, row-major, 2-byte elems) -> LDS.
// D# packing per CDNA5 ISA.
//   tile: tile_d0 elems/row x tile_d1 rows; row stride = stride0 elems.
// ---------------------------------------------------------------------------
__device__ __forceinline__ void tdm_load_2d_b16(uint32_t lds_off, const void* gaddr,
                                                uint32_t tensor_d0, uint32_t tensor_d1,
                                                uint32_t tile_d0, uint32_t tile_d1,
                                                uint32_t stride0_elems) {
    const uint64_t ga = (uint64_t)(uintptr_t)gaddr;
    // Group 0: count=1 | lds_addr | global_addr[56:0] | type=2 (bits 127:126)
    v4u g0 = { 0x1u,
               lds_off,
               (uint32_t)ga,
               (uint32_t)(ga >> 32) | 0x80000000u };
    // Group 1: data_size=1 (2B); tensor_dim0/1; tile_dim0/1/2; dim0 stride.
    v8i g1 = { (int)(1u << 16),                                 // data_size @ [17:16]
               (int)(tensor_d0 << 16),                          // tensor_dim0[15:0] @ [79:64... word1 31:16]
               (int)((tensor_d0 >> 16) | (tensor_d1 << 16)),    // dim0 hi | dim1 lo
               (int)((tensor_d1 >> 16) | (tile_d0 << 16)),      // dim1 hi | tile_dim0
               (int)(tile_d1 & 0xFFFFu),                        // tile_dim1 | tile_dim2=0
               (int)stride0_elems,                              // tensor_dim0_stride lo32
               0,                                               // stride0 hi16 | stride1 lo16
               0 };                                             // stride1 hi32
    v4i z4 = {0, 0, 0, 0};
#if defined(__clang_major__) && (__clang_major__ >= 23)
    v8i z8 = {0, 0, 0, 0, 0, 0, 0, 0};
    __builtin_amdgcn_tensor_load_to_lds(g0, g1, z4, z4, z8, 0);
#else
    __builtin_amdgcn_tensor_load_to_lds(g0, g1, z4, z4, 0);
#endif
}
#endif

// ---------------------------------------------------------------------------
// fp32 -> bf16 conversion, 4 elements/thread
// ---------------------------------------------------------------------------
__global__ __launch_bounds__(256) void cvt4(const float4* __restrict__ in,
                                            bf16_t* __restrict__ out, int n4) {
    int i = blockIdx.x * 256 + threadIdx.x;
    if (i >= n4) return;
    float4 v = in[i];
    union { bf16_t o[4]; uint2 u; } pk;
    pk.o[0] = (bf16_t)v.x; pk.o[1] = (bf16_t)v.y;
    pk.o[2] = (bf16_t)v.z; pk.o[3] = (bf16_t)v.w;
    *(uint2*)(out + (size_t)i * 4) = pk.u;
}

// ---------------------------------------------------------------------------
// In-place RoPE on bf16 tensor [B,S,nh,HD]; one thread per (even,odd) pair.
// cos/sin tables are [S, HD/2] fp32.
// ---------------------------------------------------------------------------
__global__ __launch_bounds__(256) void rope_kernel(bf16_t* __restrict__ X,
                                                   const float* __restrict__ cosT,
                                                   const float* __restrict__ sinT,
                                                   int nh, int npairs) {
    int p = blockIdx.x * 256 + threadIdx.x;
    if (p >= npairs) return;
    int d2 = p & 63;                    // HD/2 == 64
    int t  = p >> 6;                    // (b*S+s)*nh + head
    int sh = t / nh;                    // b*S + s
    int s  = sh & (S_ - 1);
    float c  = cosT[s * 64 + d2];
    float sn = sinT[s * 64 + d2];
    bf16_t* ptr = X + (size_t)p * 2;
    float x0 = (float)ptr[0];
    float x1 = (float)ptr[1];
    ptr[0] = (bf16_t)(x0 * c - x1 * sn);
    ptr[1] = (bf16_t)(x0 * sn + x1 * c);
}

// ---------------------------------------------------------------------------
// GEMM: C[M,N] = A[M,K] * W[N,K]^T   (A, W bf16 row-major; C f32 or bf16)
// Block: 256 threads = 8 waves; wave computes a 16(M) x 64(N) tile.
// grid = (N/64, M/128)
// ---------------------------------------------------------------------------
template <typename OutT>
__global__ __launch_bounds__(256) void gemm_bt(const bf16_t* __restrict__ A,
                                               const bf16_t* __restrict__ W,
                                               OutT* __restrict__ C,
                                               int M, int N, int K) {
    const int lane = threadIdx.x & 31;
    const int wave = threadIdx.x >> 5;
    const int n  = lane & 15;
    const int hh = lane >> 4;
    const int m0 = (blockIdx.y * 8 + wave) * 16;
    const int n0 = blockIdx.x * 64;

    v8f acc[4];
#pragma unroll
    for (int j = 0; j < 4; ++j) acc[j] = v8f_zero();

    for (int k0 = 0; k0 < K; k0 += 32) {
        // A fragment: lane row m = n, swizzled K offsets (8*hh .. , 16+8*hh ..)
        FragB af;
        const bf16_t* ar = A + (size_t)(m0 + n) * K + k0 + 8 * hh;
        __builtin_prefetch(ar + 128, 0, 1);     // global_prefetch_b8
        af.q[0] = *(const uint4*)ar;
        af.q[1] = *(const uint4*)(ar + 16);
#pragma unroll
        for (int j = 0; j < 4; ++j) {
            // B fragment: lane column = W row (n0+16j+n), swizzled K offsets (8*hh .. , 16+8*hh ..)
            FragB bfr;
            const bf16_t* br = W + (size_t)(n0 + 16 * j + n) * K + k0 + 8 * hh;
            __builtin_prefetch(br + 128, 0, 1);
            bfr.q[0] = *(const uint4*)br;
            bfr.q[1] = *(const uint4*)(br + 16);
            acc[j] = wmma_bf16(af.v, bfr.v, acc[j]);
        }
    }

    // C layout: VGPR r -> row m0 + 8*hh + r, column n0 + 16*j + n
#pragma unroll
    for (int j = 0; j < 4; ++j)
#pragma unroll
        for (int r = 0; r < 8; ++r)
            C[(size_t)(m0 + 8 * hh + r) * N + n0 + 16 * j + n] = (OutT)acc[j][r];
}

// ---------------------------------------------------------------------------
// Flash attention, causal, GQA (REP=4).
// Q: [B,S,H,HD] bf16 (RoPE'd)   K,V: [B,S,KV,HD] bf16 (K RoPE'd)
// O: [B,S,H,HD] bf16
// grid.x = B*H*(S/128); block = 256 threads = 8 waves; wave = 16 q rows.
// K/V tiles staged into LDS by the Tensor Data Mover (row-major [k][d]).
// ---------------------------------------------------------------------------
__global__ __launch_bounds__(256) void attn_kernel(const bf16_t* __restrict__ Q,
                                                   const bf16_t* __restrict__ K,
                                                   const bf16_t* __restrict__ V,
                                                   bf16_t* __restrict__ O) {
    __shared__ __align__(16) bf16_t ldsK[32 * 128];       // K tile [k][d]
    __shared__ __align__(16) bf16_t ldsV[32 * 128];       // V tile [k][d]
    __shared__ __align__(16) bf16_t ldsP[8][16 * 32];     // per-wave P staging [m][k]

    const int tid  = threadIdx.x;
    const int wave = tid >> 5;
    const int lane = tid & 31;
    const int n  = lane & 15;
    const int hh = lane >> 4;

    int bid = blockIdx.x;
    const int QT = S_ / 128;            // 16 q-blocks per (b,h)
    const int qt = bid % QT;  bid /= QT;
    const int h  = bid % H_;  bid /= H_;
    const int b  = bid;
    const int g  = h / REP_;            // kv head

    const int qb0 = qt * 128 + wave * 16;   // first q row (in S) of this wave

    // ---- load Q A-fragments (16 rows x 128 d) once: 4 frags of K=32 --------
    const bf16_t* Qp = Q + (((size_t)b * S_ + qb0) * H_ + h) * HD_;
    FragB qf[4];
#pragma unroll
    for (int c = 0; c < 4; ++c) {
        const bf16_t* r = Qp + (size_t)n * (H_ * HD_) + c * 32 + 8 * hh;
        qf[c].q[0] = *(const uint4*)r;
        qf[c].q[1] = *(const uint4*)(r + 16);
    }

    float mrow[8], lrow[8];
    v8f acc[8];
#pragma unroll
    for (int r = 0; r < 8; ++r) { mrow[r] = -__builtin_inff(); lrow[r] = 0.f; }
#pragma unroll
    for (int j = 0; j < 8; ++j) acc[j] = v8f_zero();

    const int nkt = (qt + 1) * 4;       // 32-wide k tiles covering causal extent

    for (int kt = 0; kt < nkt; ++kt) {
        const int k0 = kt * 32;
        __syncthreads();                // previous iteration's compute is done

        // ---- stage K and V tiles (32 rows x 128 d, row-major) --------------
        const size_t gbase = (((size_t)b * S_ + k0) * KV_ + g) * HD_;
#if USE_TDM
        if (wave == 0) {
            tdm_load_2d_b16((uint32_t)(uintptr_t)ldsK, K + gbase,
                            /*tensor_d0=*/HD_, /*tensor_d1=*/S_,
                            /*tile_d0=*/HD_,   /*tile_d1=*/32,
                            /*stride0=*/KV_ * HD_);
            tdm_load_2d_b16((uint32_t)(uintptr_t)ldsV, V + gbase,
                            HD_, S_, HD_, 32, KV_ * HD_);
            __builtin_amdgcn_s_wait_tensorcnt(0);
        }
#else
        {
            const int row = tid >> 3;           // 0..31 (k within tile)
            const int col = (tid & 7) * 16;     // 0..112 (d)
            const size_t gidx = gbase + (size_t)row * (KV_ * HD_) + col;
            bf16_t* kd = ldsK + row * 128 + col;
            *(uint4*)kd       = *(const uint4*)(K + gidx);
            *(uint4*)(kd + 8) = *(const uint4*)(K + gidx + 8);
            bf16_t* vd = ldsV + row * 128 + col;
            *(uint4*)vd       = *(const uint4*)(V + gidx);
            *(uint4*)(vd + 8) = *(const uint4*)(V + gidx + 8);
        }
#endif
        __syncthreads();

        if (k0 <= qb0 + 15) {           // tile intersects this wave's causal region
            // ---- S = Q K^T : two 16x16 f32 tiles over HD=128 ---------------
            v8f s0 = v8f_zero(), s1 = v8f_zero();
#pragma unroll
            for (int c = 0; c < 4; ++c) {
                FragB kf0, kf1;
                const bf16_t* p0 = ldsK + (size_t)(0  + n) * 128 + c * 32 + 8 * hh;
                const bf16_t* p1 = ldsK + (size_t)(16 + n) * 128 + c * 32 + 8 * hh;
                kf0.q[0] = *(const uint4*)p0; kf0.q[1] = *(const uint4*)(p0 + 16);
                kf1.q[0] = *(const uint4*)p1; kf1.q[1] = *(const uint4*)(p1 + 16);
                s0 = wmma_bf16(qf[c].v, kf0.v, s0);
                s1 = wmma_bf16(qf[c].v, kf1.v, s1);
            }

            // ---- masked online softmax (row = qb0+8*hh+r, col = k0+16t+n) --
            float alpha[8];
#pragma unroll
            for (int r = 0; r < 8; ++r) {
                const int qrow = qb0 + 8 * hh + r;
                float a = s0[r] * SCALE_;
                float bvl = s1[r] * SCALE_;
                if (k0 + n      > qrow) a   = -__builtin_inff();
                if (k0 + 16 + n > qrow) bvl = -__builtin_inff();
                s0[r] = a; s1[r] = bvl;
                float mx = fmaxf(a, bvl);
                mx = fmaxf(mx, __shfl_xor(mx, 1, 32));
                mx = fmaxf(mx, __shfl_xor(mx, 2, 32));
                mx = fmaxf(mx, __shfl_xor(mx, 4, 32));
                mx = fmaxf(mx, __shfl_xor(mx, 8, 32));
                const float mn = fmaxf(mrow[r], mx);
                alpha[r] = __expf(mrow[r] - mn);
                mrow[r] = mn;
                float p0e = __expf(s0[r] - mn);
                float p1e = __expf(s1[r] - mn);
                s0[r] = p0e; s1[r] = p1e;
                float rs = p0e + p1e;
                rs += __shfl_xor(rs, 1, 32);
                rs += __shfl_xor(rs, 2, 32);
                rs += __shfl_xor(rs, 4, 32);
                rs += __shfl_xor(rs, 8, 32);
                lrow[r] = lrow[r] * alpha[r] + rs;
            }
#pragma unroll
            for (int j = 0; j < 8; ++j)
#pragma unroll
                for (int r = 0; r < 8; ++r) acc[j][r] *= alpha[r];

            // ---- restage P (C-layout -> A-layout) through LDS --------------
            bf16_t* Pw = &ldsP[wave][0];
#pragma unroll
            for (int r = 0; r < 8; ++r) {
                Pw[(8 * hh + r) * 32 + n]      = (bf16_t)s0[r];
                Pw[(8 * hh + r) * 32 + 16 + n] = (bf16_t)s1[r];
            }
            __builtin_amdgcn_wave_barrier();   // DS is in-order per wave; block reordering

            FragB pf;
            const bf16_t* pr = Pw + (size_t)n * 32;
            pf.q[0] = *(const uint4*)(pr + 16 * hh);
            pf.q[1] = *(const uint4*)(pr + 16 * hh + 8);

            // ---- O += P * V : B-fragment gathered from row-major V tile ----
            //      lane column d = 16j+n, elements k = 16*hh + e
            const bf16_t* vb = ldsV + (size_t)hh * 16 * 128 + n;
#pragma unroll
            for (int j = 0; j < 8; ++j) {
                FragB vf;
#pragma unroll
                for (int e = 0; e < 16; ++e)
                    vf.e[e] = vb[e * 128 + 16 * j];
                acc[j] = wmma_bf16(pf.v, vf.v, acc[j]);
            }
        }
    }

    // ---- normalize and write out [B,S,H,HD] bf16 ---------------------------
    bf16_t* Op = O + (((size_t)b * S_ + qb0) * H_ + h) * HD_;
#pragma unroll
    for (int j = 0; j < 8; ++j)
#pragma unroll
        for (int r = 0; r < 8; ++r) {
            const float val = acc[j][r] / lrow[r];
            Op[(size_t)(8 * hh + r) * (H_ * HD_) + 16 * j + n] = (bf16_t)val;
        }
}

// ---------------------------------------------------------------------------
// Host-side launcher
// ---------------------------------------------------------------------------
extern "C" void kernel_launch(void* const* d_in, const int* in_sizes, int n_in,
                              void* d_out, int out_size, void* d_ws, size_t ws_size,
                              hipStream_t stream) {
    const float* x  = (const float*)d_in[0];
    const float* fc = (const float*)d_in[1];
    const float* fs = (const float*)d_in[2];
    const float* wq = (const float*)d_in[3];
    const float* wk = (const float*)d_in[4];
    const float* wv = (const float*)d_in[5];
    const float* wo = (const float*)d_in[6];
    float* out = (float*)d_out;

    char* p = (char*)d_ws;
    auto alloc = [&](size_t bytes) {
        void* r = (void*)p;
        p += (bytes + 255) & ~(size_t)255;
        return r;
    };
    bf16_t* Xb  = (bf16_t*)alloc((size_t)BS_ * D_ * 2);
    bf16_t* Wqb = (bf16_t*)alloc((size_t)D_ * D_ * 2);
    bf16_t* Wkb = (bf16_t*)alloc((size_t)KD_ * D_ * 2);
    bf16_t* Wvb = (bf16_t*)alloc((size_t)KD_ * D_ * 2);
    bf16_t* Wob = (bf16_t*)alloc((size_t)D_ * D_ * 2);
    bf16_t* Qb  = (bf16_t*)alloc((size_t)BS_ * D_ * 2);
    bf16_t* Kb  = (bf16_t*)alloc((size_t)BS_ * KD_ * 2);
    bf16_t* Vb  = (bf16_t*)alloc((size_t)BS_ * KD_ * 2);
    bf16_t* Ab  = (bf16_t*)alloc((size_t)BS_ * D_ * 2);

    // --- fp32 -> bf16 conversions ---
    {
        int n4;
        n4 = BS_ * D_ / 4;  cvt4<<<dim3((n4 + 255) / 256), 256, 0, stream>>>((const float4*)x,  Xb,  n4);
        n4 = D_ * D_ / 4;   cvt4<<<dim3((n4 + 255) / 256), 256, 0, stream>>>((const float4*)wq, Wqb, n4);
        n4 = KD_ * D_ / 4;  cvt4<<<dim3((n4 + 255) / 256), 256, 0, stream>>>((const float4*)wk, Wkb, n4);
        n4 = KD_ * D_ / 4;  cvt4<<<dim3((n4 + 255) / 256), 256, 0, stream>>>((const float4*)wv, Wvb, n4);
        n4 = D_ * D_ / 4;   cvt4<<<dim3((n4 + 255) / 256), 256, 0, stream>>>((const float4*)wo, Wob, n4);
    }

    // --- QKV projections (bf16 WMMA, bf16 out) ---
    gemm_bt<bf16_t><<<dim3(D_ / 64,  BS_ / 128), 256, 0, stream>>>(Xb, Wqb, Qb, BS_, D_,  D_);
    gemm_bt<bf16_t><<<dim3(KD_ / 64, BS_ / 128), 256, 0, stream>>>(Xb, Wkb, Kb, BS_, KD_, D_);
    gemm_bt<bf16_t><<<dim3(KD_ / 64, BS_ / 128), 256, 0, stream>>>(Xb, Wvb, Vb, BS_, KD_, D_);

    // --- RoPE on Q and K (in place) ---
    {
        int npq = BS_ * H_ * (HD_ / 2);   // 8,388,608
        int npk = BS_ * KV_ * (HD_ / 2);  // 2,097,152
        rope_kernel<<<dim3((npq + 255) / 256), 256, 0, stream>>>(Qb, fc, fs, H_,  npq);
        rope_kernel<<<dim3((npk + 255) / 256), 256, 0, stream>>>(Kb, fc, fs, KV_, npk);
    }

    // --- causal GQA flash attention ---
    attn_kernel<<<dim3(B_ * H_ * (S_ / 128)), 256, 0, stream>>>(Qb, Kb, Vb, Ab);

    // --- output projection (fp32 out) ---
    gemm_bt<float><<<dim3(D_ / 64, BS_ / 128), 256, 0, stream>>>(Ab, Wob, out, BS_, D_, D_);
}
